// TriMulUpdate_22522808500393
// MI455X (gfx1250) — matrix operand order repaired and checked by reading, not compiled, not run on hardware
//
#include <hip/hip_runtime.h>
#include <stdint.h>

#define Nn 384
#define Dc 128
#define NN (Nn * Nn)

typedef __attribute__((ext_vector_type(16))) __bf16 v16bf;
typedef __attribute__((ext_vector_type(8)))  __bf16 v8bf;
typedef __attribute__((ext_vector_type(8)))  float  v8f;
typedef __attribute__((ext_vector_type(4)))  float  v4f;

__device__ __forceinline__ uint16_t f32_to_bf16(float f) {
  uint32_t u = __builtin_bit_cast(uint32_t, f);
  uint32_t r = u + 0x7FFFu + ((u >> 16) & 1u);   // round-to-nearest-even
  return (uint16_t)(r >> 16);
}

__device__ __forceinline__ v16bf cat8(v8bf lo, v8bf hi) {
  return __builtin_shufflevector(lo, hi, 0, 1, 2, 3, 4, 5, 6, 7,
                                         8, 9, 10, 11, 12, 13, 14, 15);
}

// ---------------------------------------------------------------------------
// Kernel A: fused linear.  grid = (384, 2).
//   mat 0 (row walk):    Lbuf[d][i][k] = pair[i,k,:] . Wa[d,:] + ba[d]
//   mat 1 (column walk): Rt  [d][j][k] = pair[k,j,:] . Wb[d,:] + bb[d]
// Both outputs are k-contiguous, so kernel B needs no transpose anywhere.
// ---------------------------------------------------------------------------
__global__ __launch_bounds__(256) void trimul_linear(
    const float* __restrict__ pairp,
    const float* __restrict__ Wa, const float* __restrict__ ba,
    const float* __restrict__ Wb, const float* __restrict__ bb,
    uint16_t* __restrict__ ws)
{
  const int r   = blockIdx.x;     // i (left) or j (right)
  const int mat = blockIdx.y;     // 0 = left, 1 = right
  const float* W    = mat ? Wb : Wa;
  const float* bias = mat ? bb : ba;
  uint16_t* obuf = ws + (size_t)mat * (size_t)Dc * NN;

  // element (c, d) of this WG's slab lives at  pbase + c*cstride + d
  const size_t cstride = mat ? (size_t)Nn * Dc : (size_t)Dc;
  const float* pbase   = pairp + (mat ? (size_t)r * Dc : (size_t)r * Nn * Dc);

  const int t = threadIdx.x;

  __shared__ float Wl[64 * 132];   // half of W (64 e rows), padded
  __shared__ float Sl[32 * 132];   // 32-column slab, padded

  const int el = t >> 3;          // 0..31
  const int cl = (t & 7) * 4;     // 0..28

  for (int eph = 0; eph < 2; ++eph) {
    for (int idx = t; idx < 64 * 32; idx += 256) {
      int e  = idx >> 5;
      int d4 = idx & 31;
      v4f w = *(const v4f*)(W + (size_t)(eph * 64 + e) * Dc + d4 * 4);
      *(v4f*)(&Wl[e * 132 + d4 * 4]) = w;
    }
    __syncthreads();

    for (int cb = 0; cb < 12; ++cb) {
      for (int idx = t; idx < 32 * 32; idx += 256) {
        int c  = idx >> 5;
        int d4 = idx & 31;
        v4f s = *(const v4f*)(pbase + (size_t)(cb * 32 + c) * cstride + d4 * 4);
        *(v4f*)(&Sl[c * 132 + d4 * 4]) = s;
      }
      __syncthreads();

      float acc[2][4];
      #pragma unroll
      for (int eo = 0; eo < 2; ++eo)
        #pragma unroll
        for (int cu = 0; cu < 4; ++cu) acc[eo][cu] = 0.0f;

      #pragma unroll 4
      for (int d4 = 0; d4 < 32; ++d4) {
        v4f w0 = *(const v4f*)(&Wl[(el     ) * 132 + d4 * 4]);
        v4f w1 = *(const v4f*)(&Wl[(el + 32) * 132 + d4 * 4]);
        v4f sc[4];
        #pragma unroll
        for (int cu = 0; cu < 4; ++cu)
          sc[cu] = *(const v4f*)(&Sl[(cl + cu) * 132 + d4 * 4]);
        #pragma unroll
        for (int cu = 0; cu < 4; ++cu)
          #pragma unroll
          for (int q = 0; q < 4; ++q) {
            acc[0][cu] += w0[q] * sc[cu][q];
            acc[1][cu] += w1[q] * sc[cu][q];
          }
      }

      #pragma unroll
      for (int eo = 0; eo < 2; ++eo) {
        int e = eph * 64 + eo * 32 + el;
        float bv = bias[e];
        uint16_t o[4];
        #pragma unroll
        for (int cu = 0; cu < 4; ++cu) o[cu] = f32_to_bf16(acc[eo][cu] + bv);
        size_t base = (size_t)e * NN + (size_t)r * Nn + (size_t)(cb * 32 + cl);
        uint2 pv;
        pv.x = (uint32_t)o[0] | ((uint32_t)o[1] << 16);
        pv.y = (uint32_t)o[2] | ((uint32_t)o[3] << 16);
        *(uint2*)(obuf + base) = pv;
      }
      __syncthreads();
    }
    __syncthreads();
  }
}

// ---------------------------------------------------------------------------
// Kernel B: upd[i,j,d] = sum_k L[d][i][k] * Rt[d][j][k], out = pair + upd.
// grid = (6 i-tiles, 6 j-tiles, 16 channel groups); 8 waves, one channel per
// wave, wave tile 64x64 (4x4 WMMA frags).  All fragment data comes straight
// from global (L2-resident bf16), no LDS / barriers in the k-loop.
// Epilogue stages C through LDS so stores are 32B channel-contiguous.
// ---------------------------------------------------------------------------
__global__ __launch_bounds__(256) void trimul_mm(
    const uint16_t* __restrict__ ws,
    const float* __restrict__ pairp,
    float* __restrict__ outp)
{
  const int i0 = blockIdx.x * 64;
  const int j0 = blockIdx.y * 64;
  const int d0 = blockIdx.z * 8;

  const int t    = threadIdx.x;
  const int lane = t & 31;
  const int wv   = t >> 5;        // wave = channel within group
  const int d    = d0 + wv;

  const uint16_t* La = ws + (size_t)d * NN;                    // [i][k]
  const uint16_t* Rt = ws + (size_t)Dc * NN + (size_t)d * NN;  // [j][k]

  const int nl = lane & 15;
  // 16-bit A fragment: lane half h=0/8, k in {h..h+7} U {h+16..h+23}
  const int hA = (lane < 16) ? 0 : 8;
  // B fragment: lane n = lane&15, k in {h..h+7} U {h+16..h+23}, h=0/8
  const int hB = (lane < 16) ? 0 : 8;

  v8f acc[4][4];
  const v8f vzero = {0.f, 0.f, 0.f, 0.f, 0.f, 0.f, 0.f, 0.f};
  #pragma unroll
  for (int ai = 0; ai < 4; ++ai)
    #pragma unroll
    for (int bj = 0; bj < 4; ++bj) acc[ai][bj] = vzero;

  const uint16_t* aBase = La + (size_t)(i0 + nl) * Nn + hA;
  const uint16_t* bBase = Rt + (size_t)(j0 + nl) * Nn + hB;

  for (int k0 = 0; k0 < Nn; k0 += 32) {
    v16bf bfr[4];
    #pragma unroll
    for (int bj = 0; bj < 4; ++bj) {
      const uint16_t* p = bBase + (size_t)(bj * 16) * Nn + k0;
      v8bf lo = *(const v8bf*)(p);
      v8bf hi = *(const v8bf*)(p + 16);
      bfr[bj] = cat8(lo, hi);
    }
    #pragma unroll
    for (int ai = 0; ai < 4; ++ai) {
      const uint16_t* p = aBase + (size_t)(ai * 16) * Nn + k0;
      v8bf lo = *(const v8bf*)(p);
      v8bf hi = *(const v8bf*)(p + 16);
      v16bf af = cat8(lo, hi);
      #pragma unroll
      for (int bj = 0; bj < 4; ++bj)
        acc[ai][bj] = __builtin_amdgcn_wmma_f32_16x16x32_bf16(
            false, af, false, bfr[bj], (short)0, acc[ai][bj], false, false);
    }
  }

  // -------- epilogue: channel-interleaved staging for coalesced stores -----
  // Cl layout: [m(16)][n(64)][ch(8)] floats = 32KB.  Fully unrolled so that
  // every accumulator index is a compile-time constant (no scratch spills).
  __shared__ float Cl[16 * 64 * 8];
  const int mo = (lane >> 4) << 3;     // C layout: row m = v + mo
  const int en = t & 63;               // epilogue n
  const int em = t >> 6;               // epilogue m base (0..3)

  #pragma unroll
  for (int ai = 0; ai < 4; ++ai) {
    #pragma unroll
    for (int bj = 0; bj < 4; ++bj)
      #pragma unroll
      for (int v = 0; v < 8; ++v)
        Cl[((v + mo) * 64 + bj * 16 + nl) * 8 + wv] = acc[ai][bj][v];
    __syncthreads();

    #pragma unroll
    for (int ms = 0; ms < 4; ++ms) {
      int m  = em * 4 + ms;
      int gi = i0 + ai * 16 + m;
      int gj = j0 + en;
      size_t idx = ((size_t)gi * Nn + (size_t)gj) * Dc + (size_t)d0;
      v4f p0 = *(const v4f*)(pairp + idx);
      v4f p1 = *(const v4f*)(pairp + idx + 4);
      const float* cs = &Cl[(m * 64 + en) * 8];
      v4f c0 = *(const v4f*)(cs);
      v4f c1 = *(const v4f*)(cs + 4);
      v4f r0, r1;
      #pragma unroll
      for (int q = 0; q < 4; ++q) { r0[q] = p0[q] + c0[q]; r1[q] = p1[q] + c1[q]; }
      *(v4f*)(outp + idx)     = r0;
      *(v4f*)(outp + idx + 4) = r1;
    }
    __syncthreads();
  }
}

// ---------------------------------------------------------------------------
extern "C" void kernel_launch(void* const* d_in, const int* in_sizes, int n_in,
                              void* d_out, int out_size, void* d_ws, size_t ws_size,
                              hipStream_t stream) {
  const float* pairp = (const float*)d_in[0];
  const float* Wa    = (const float*)d_in[1];
  const float* ba    = (const float*)d_in[2];
  const float* Wb    = (const float*)d_in[3];
  const float* bb    = (const float*)d_in[4];
  float* outp  = (float*)d_out;
  uint16_t* ws = (uint16_t*)d_ws;   // Lbuf (37.75MB) + Rt (37.75MB), bf16

  trimul_linear<<<dim3(Nn, 2, 1), 256, 0, stream>>>(pairp, Wa, ba, Wb, bb, ws);
  trimul_mm<<<dim3(6, 6, 16), 256, 0, stream>>>(ws, pairp, outp);
}
